// OODDetectionHead_83708912599141
// MI455X (gfx1250) — matrix operand order repaired and checked by reading, not compiled, not run on hardware
//
#include <hip/hip_runtime.h>
#include <hip/hip_bf16.h>

typedef __attribute__((ext_vector_type(16))) _Float16 v16h;
typedef __attribute__((ext_vector_type(8)))  _Float16 v8h;
typedef __attribute__((ext_vector_type(8)))  float    v8f;

#define EPSF    1e-7f
#define B_ROWS  4096
#define L_ROWS  50000
#define DIM     256
#define NTILES  (L_ROWS / 16)   // 3125

union Frag { v16h v; v8h h[2]; };

// ---------------------------------------------------------------------------
// Prep: f32 rows -> f16 copy, plus ||row||^2 and 1/max(1-||row||^2, EPS).
// One wave (32 lanes) per row; 8 waves per 256-thread block.
// ---------------------------------------------------------------------------
__global__ void prep_rows_kernel(const float* __restrict__ src,
                                 _Float16* __restrict__ dst,
                                 float* __restrict__ n2out,
                                 float* __restrict__ invout,
                                 int nrows) {
  int row = blockIdx.x * 8 + (threadIdx.x >> 5);
  if (row >= nrows) return;
  int lane = threadIdx.x & 31;
  const float* p = src + (size_t)row * DIM;
  _Float16* q = dst + (size_t)row * DIM;
  float s = 0.0f;
#pragma unroll
  for (int j = 0; j < 8; ++j) {
    float v = p[lane + 32 * j];
    s = fmaf(v, v, s);
    q[lane + 32 * j] = (_Float16)v;
  }
#pragma unroll
  for (int off = 16; off >= 1; off >>= 1) s += __shfl_xor(s, off, 32);
  if (lane == 0) {
    n2out[row] = s;
    invout[row] = 1.0f / fmaxf(1.0f - s, EPSF);
  }
}

// ---------------------------------------------------------------------------
// Main: per 16-row tile of z, scan all 50000 leaves with WMMA f16 GEMM,
// rank by monotone arccosh argument, keep per-lane top-5, merge at the end.
// Block = 256 threads = 8 waves; wave w owns column tiles w, w+8, w+16, ...
// ---------------------------------------------------------------------------
__global__ void __launch_bounds__(256, 1)
ood_wmma_kernel(const _Float16* __restrict__ z16,
                const _Float16* __restrict__ leaf16,
                const float* __restrict__ x2,  const float* __restrict__ invrx,
                const float* __restrict__ y2,  const float* __restrict__ invry,
                const int* __restrict__ leaf_ids,
                const float* __restrict__ thr_p,
                float* __restrict__ out) {
  __shared__ float sTile[8][16][16];   // per-wave 16x16 arg tile
  __shared__ float sArg[8][32][5];     // per-lane top5 values
  __shared__ int   sIdx[8][32][5];     // per-lane top5 indices

  const int wave = threadIdx.x >> 5;
  const int lane = threadIdx.x & 31;
  const int m    = lane & 15;          // row within tile (A) / col within tile (B)
  const int hi   = lane >> 4;
  const int rowBase = blockIdx.x * 16;

  // Resident A fragments: 16 rows x 256 K, ISA 16-bit A layout.
  // lanes 0-15: K = {k0..k0+7, k0+16..k0+23}; lanes 16-31: +8.
  Frag af[8];
#pragma unroll
  for (int s = 0; s < 8; ++s) {
    const _Float16* pa = z16 + (size_t)(rowBase + m) * DIM + 32 * s + 8 * hi;
    af[s].h[0] = *(const v8h*)(pa);
    af[s].h[1] = *(const v8h*)(pa + 16);
  }

  // Per-lane row constants in WMMA C layout: VGPR v -> row v + 8*hi.
  float xs2[8], sc2[8];
#pragma unroll
  for (int v = 0; v < 8; ++v) {
    int r = rowBase + v + 8 * hi;
    xs2[v] = x2[r];
    sc2[v] = 2.0f * invrx[r];
  }

  // top-5 (ascending) in scalar registers only.
  float t0 = 3.4e38f, t1 = 3.4e38f, t2v = 3.4e38f, t3 = 3.4e38f, t4 = 3.4e38f;
  int   i0 = 0, i1 = 0, i2v = 0, i3 = 0, i4 = 0;

  for (int nt = wave; nt < NTILES; nt += 8) {
    const int nbase = nt * 16;
    const int col   = nbase + m;
    // ISA 16-bit B layout: lane's column = m, lanes 0-15 K={k0..k0+7, k0+16..k0+23},
    // lanes 16-31: +8 (matches A).
    const _Float16* pb = leaf16 + (size_t)col * DIM + 8 * hi;

    // ---- load phase: batch ALL tile loads so they land in one clause ----
    Frag bf[8];
#pragma unroll
    for (int s = 0; s < 8; ++s) {
      bf[s].h[0] = *(const v8h*)(pb + 32 * s);
      bf[s].h[1] = *(const v8h*)(pb + 32 * s + 16);
    }
    const float y2l = y2[col];
    const float scy = invry[col];

    // Speculative prefetch 8 tiles ahead (no clamp: OOB prefetch is silently
    // dropped per ISA, and col+128 stays inside the workspace allocation).
    __builtin_prefetch(leaf16 + (size_t)(col + 128) * DIM, 0, 3);

    // ---- compute phase: two independent WMMA accumulation chains ----
    v8f acc0 = {};
    v8f acc1 = {};
#pragma unroll
    for (int s = 0; s < 8; s += 2) {
      acc0 = __builtin_amdgcn_wmma_f32_16x16x32_f16(false, af[s].v, false, bf[s].v,
                                                    (short)0, acc0, false, false);
      acc1 = __builtin_amdgcn_wmma_f32_16x16x32_f16(false, af[s + 1].v, false, bf[s + 1].v,
                                                    (short)0, acc1, false, false);
    }

#pragma unroll
    for (int v = 0; v < 8; ++v) {
      float xyv = acc0[v] + acc1[v];
      float sq  = fmaxf(xs2[v] + y2l - 2.0f * xyv, 0.0f);
      float arg = fmaf(sq, sc2[v] * scy, 1.0f);
      arg = fmaxf(arg, 1.0f + EPSF);
      sTile[wave][v + 8 * hi][m] = arg;   // row-major in LDS
    }
    __builtin_amdgcn_sched_barrier(0);    // intra-wave: DS ops are in-order

    // Re-read row-wise: lane handles row m, columns 8*hi .. 8*hi+7.
#pragma unroll
    for (int j = 0; j < 8; ++j) {
      float v = sTile[wave][m][8 * hi + j];
      int gcol = nbase + 8 * hi + j;
      if (v < t4) {
        if (v < t3) {
          t4 = t3; i4 = i3;
          if (v < t2v) {
            t3 = t2v; i3 = i2v;
            if (v < t1) {
              t2v = t1; i2v = i1;
              if (v < t0) { t1 = t0; i1 = i0; t0 = v; i0 = gcol; }
              else        { t1 = v;  i1 = gcol; }
            } else { t2v = v; i2v = gcol; }
          } else { t3 = v; i3 = gcol; }
        } else { t4 = v; i4 = gcol; }
      }
    }
    __builtin_amdgcn_sched_barrier(0);
  }

  sArg[wave][lane][0] = t0;  sIdx[wave][lane][0] = i0;
  sArg[wave][lane][1] = t1;  sIdx[wave][lane][1] = i1;
  sArg[wave][lane][2] = t2v; sIdx[wave][lane][2] = i2v;
  sArg[wave][lane][3] = t3;  sIdx[wave][lane][3] = i3;
  sArg[wave][lane][4] = t4;  sIdx[wave][lane][4] = i4;
  __syncthreads();

  // Merge 16 candidate lists (8 waves x 2 half-lanes) per row; thread r = row r.
  if (threadIdx.x < 16) {
    const int r = threadIdx.x;
    const float thr = *thr_p;
    int picked[5];
#pragma unroll
    for (int k = 0; k < 5; ++k) picked[k] = -1;
    float score = 0.0f;
#pragma unroll
    for (int pick = 0; pick < 5; ++pick) {
      float mv = 3.5e38f; int mi = 0; int mpos = -1;
      for (int w = 0; w < 8; ++w) {
        for (int h = 0; h < 2; ++h) {
          int ln = r + 16 * h;
          for (int k = 0; k < 5; ++k) {
            int pos = ((w * 32) + ln) * 5 + k;
            bool used = false;
#pragma unroll
            for (int j = 0; j < 5; ++j) used |= (picked[j] == pos);
            float v = sArg[w][ln][k];
            if (!used && v < mv) { mv = v; mi = sIdx[w][ln][k]; mpos = pos; }
          }
        }
      }
      picked[pick] = mpos;
      // arccosh only for the winners
      float d = logf(mv + sqrtf(fmaxf(fmaf(mv, mv, -1.0f), 0.0f)));
      if (pick == 0) score = d;
      out[2 * B_ROWS + (size_t)(rowBase + r) * 5 + pick] = (float)leaf_ids[mi];
    }
    out[rowBase + r] = score;
    out[B_ROWS + rowBase + r] = (score > thr) ? 1.0f : 0.0f;
  }
}

// ---------------------------------------------------------------------------
extern "C" void kernel_launch(void* const* d_in, const int* in_sizes, int n_in,
                              void* d_out, int out_size, void* d_ws, size_t ws_size,
                              hipStream_t stream) {
  (void)in_sizes; (void)n_in; (void)out_size; (void)ws_size;
  const float* z    = (const float*)d_in[0];
  const float* leaf = (const float*)d_in[1];
  const int*   ids  = (const int*)d_in[2];
  const float* thr  = (const float*)d_in[3];
  float* out = (float*)d_out;

  // Workspace layout (all offsets 16B-aligned):
  char* ws = (char*)d_ws;
  _Float16* leaf16 = (_Float16*)(ws);              // 50000*256*2 = 25,600,000
  _Float16* z16    = (_Float16*)(ws + 25600000);   //  4096*256*2 =  2,097,152
  float* y2    = (float*)(ws + 27697152);          // 50000*4
  float* invry = (float*)(ws + 27897152);          // 50000*4
  float* x2    = (float*)(ws + 28097152);          //  4096*4
  float* invrx = (float*)(ws + 28113536);          //  4096*4

  prep_rows_kernel<<<(L_ROWS + 7) / 8, 256, 0, stream>>>(leaf, leaf16, y2, invry, L_ROWS);
  prep_rows_kernel<<<(B_ROWS + 7) / 8, 256, 0, stream>>>(z, z16, x2, invrx, B_ROWS);
  ood_wmma_kernel<<<B_ROWS / 16, 256, 0, stream>>>(z16, leaf16, x2, invrx,
                                                   y2, invry, ids, thr, out);
}
